// DWMF_51874615001833
// MI455X (gfx1250) — matrix operand order repaired and checked by reading, not compiled, not run on hardware
//
#include <hip/hip_runtime.h>
#include <hip/hip_bf16.h>
#include <math.h>

typedef __bf16 bf16;
typedef __attribute__((ext_vector_type(16))) __bf16 v16bf;
typedef __attribute__((ext_vector_type(8)))  float  v8f;

union Frag {
  v16bf v;
  uint4 u[2];
};

#define TWO_PI 6.28318530717958647692f

// ---------------------------------------------------------------------------
// Sine positional embedding table: pos[s*256 + c], s = py*20+px, c = channel.
// ---------------------------------------------------------------------------
__global__ void pos_kernel(float* __restrict__ pos) {
  int idx = blockIdx.x * 256 + threadIdx.x;           // 400*256 = 102400
  int s = idx >> 8, c = idx & 255;
  int py = s / 20, px = s % 20;
  float coord = (c < 128) ? ((float)py + 0.5f) : ((float)px + 0.5f);
  coord = coord / (20.0f + 1e-6f) * TWO_PI;
  int f = (c < 128) ? c : (c - 128);
  float dt = __powf(10000.0f, (2.0f * (float)(f >> 1)) / 128.0f);
  float ang = coord / dt;
  pos[idx] = (f & 1) ? __cosf(ang) : __sinf(ang);
}

// ---------------------------------------------------------------------------
// Bilinear upsample of batch-4 (global) image 80x80 -> 160x160, half-pixel.
// ---------------------------------------------------------------------------
__global__ void upglb_kernel(const float* __restrict__ x, float* __restrict__ upg) {
  int idx = blockIdx.x * 256 + threadIdx.x;           // 256*160*160
  int c = idx / 25600;
  int rem = idx % 25600;
  int Y = rem / 160, X = rem % 160;
  const float* g = x + (size_t)4 * 256 * 6400 + (size_t)c * 6400;
  float sy = 0.5f * (float)Y - 0.25f;
  float sx = 0.5f * (float)X - 0.25f;
  int y0 = (int)floorf(sy); float wy = sy - (float)y0;
  int x0 = (int)floorf(sx); float wx = sx - (float)x0;
  int y1 = min(y0 + 1, 79); y0 = max(y0, 0);
  int x1 = min(x0 + 1, 79); x0 = max(x0, 0);
  float v00 = g[y0 * 80 + x0], v01 = g[y0 * 80 + x1];
  float v10 = g[y1 * 80 + x0], v11 = g[y1 * 80 + x1];
  upg[idx] = (1.0f - wy) * ((1.0f - wx) * v00 + wx * v01)
           + wy * ((1.0f - wx) * v10 + wx * v11);
}

// ---------------------------------------------------------------------------
// Convert Wq (rows 0..255) and Wk (rows 256..511) of in_proj_w to bf16.
// ---------------------------------------------------------------------------
__global__ void wcvt_kernel(const float* __restrict__ w, bf16* __restrict__ wqk) {
  int idx = blockIdx.x * 256 + threadIdx.x;           // 512*256
  wqk[idx] = (bf16)w[idx];
}

// ---------------------------------------------------------------------------
// Build bf16 row-major [25600, 256] query input (loc_p + pos) and key input
// (patched upsampled global + pos).
// ---------------------------------------------------------------------------
__global__ void qkin_kernel(const float* __restrict__ x, const float* __restrict__ upg,
                            const float* __restrict__ pos,
                            bf16* __restrict__ qin, bf16* __restrict__ kin) {
  int idx = blockIdx.x * 256 + threadIdx.x;           // 25600*256
  int row = idx >> 8, c = idx & 255;
  int patch = row / 400, s = row % 400;
  int py = s / 20, px = s % 20;
  float p = pos[s * 256 + c];
  int b = patch >> 4, r2 = patch & 15, i = r2 >> 2, j = r2 & 3;
  float qv = x[((size_t)b * 256 + c) * 6400 + (i * 20 + py) * 80 + (j * 20 + px)] + p;
  qin[idx] = (bf16)qv;
  int i8 = patch >> 3, j8 = patch & 7;
  float kv = upg[(size_t)c * 25600 + (i8 * 20 + py) * 160 + (j8 * 20 + px)] + p;
  kin[idx] = (bf16)kv;
}

// ---------------------------------------------------------------------------
// Projection GEMM: Out[25600,256] = In[25600,256] @ W^T + bias, (*inv for Q).
// One wave per 16x64 output slab, explicit double buffering: the next
// K-chunk's A + 4xB fragments are loaded while the current chunk's 4 WMMAs
// execute, so waits before WMMAs are partial instead of full drains.
// ---------------------------------------------------------------------------
__device__ __forceinline__ void load_chunk(const bf16* arow, const bf16* brow0,
                                           const bf16* brow1, const bf16* brow2,
                                           const bf16* brow3, int kc, int half,
                                           Frag& a, Frag& b0, Frag& b1,
                                           Frag& b2, Frag& b3) {
  int ka = kc * 32 + half * 16;
  int kb = kc * 32 + half * 16;
  const uint4* ap  = (const uint4*)(arow  + ka);
  const uint4* bp0 = (const uint4*)(brow0 + kb);
  const uint4* bp1 = (const uint4*)(brow1 + kb);
  const uint4* bp2 = (const uint4*)(brow2 + kb);
  const uint4* bp3 = (const uint4*)(brow3 + kb);
  a.u[0]  = ap[0];  a.u[1]  = ap[1];   // 16 contiguous bf16
  b0.u[0] = bp0[0]; b0.u[1] = bp0[1];  // 16 contiguous bf16
  b1.u[0] = bp1[0]; b1.u[1] = bp1[1];
  b2.u[0] = bp2[0]; b2.u[1] = bp2[1];
  b3.u[0] = bp3[0]; b3.u[1] = bp3[1];
}

__global__ void __launch_bounds__(128, 1) proj_kernel(
    const bf16* __restrict__ qin, const bf16* __restrict__ kin,
    const bf16* __restrict__ wqk, const float* __restrict__ bvec,
    bf16* __restrict__ Qp, bf16* __restrict__ Kp) {
  int lane = threadIdx.x & 31, wave = threadIdx.x >> 5;
  int mtile = blockIdx.y * 4 + wave;    // 0..1599
  int ngrp  = blockIdx.x;               // 0..3  (4 consecutive 16-col N tiles)
  int which = blockIdx.z;               // 0 = Q, 1 = K
  const bf16* in    = which ? kin : qin;
  const bf16* W     = wqk + (size_t)which * 256 * 256;
  const float* bias = bvec + which * 256;
  bf16* out = which ? Kp : Qp;
  float scale = which ? 1.0f : 0.17677669529663687f;  // 1/sqrt(32) on Q only

  int half = lane >> 4, mr = lane & 15;
  // A (16x32 bf16): lane mr holds k = half*16 + e, e = 0..15
  const bf16* arow = in + (size_t)(mtile * 16 + mr) * 256;
  // B (32x16 bf16): lane mr is column, holds k = half*16 + e, e = 0..15
  const bf16* brow0 = W + (size_t)(ngrp * 64 + mr) * 256;
  const bf16* brow1 = brow0 + (size_t)16 * 256;
  const bf16* brow2 = brow0 + (size_t)32 * 256;
  const bf16* brow3 = brow0 + (size_t)48 * 256;

  v8f acc0 = {}, acc1 = {}, acc2 = {}, acc3 = {};
  Frag a[2], b0[2], b1[2], b2[2], b3[2];
  load_chunk(arow, brow0, brow1, brow2, brow3, 0, half,
             a[0], b0[0], b1[0], b2[0], b3[0]);
#pragma unroll
  for (int kc = 0; kc < 8; ++kc) {
    int cur = kc & 1, nxt = cur ^ 1;
    if (kc < 7)
      load_chunk(arow, brow0, brow1, brow2, brow3, kc + 1, half,
                 a[nxt], b0[nxt], b1[nxt], b2[nxt], b3[nxt]);
    acc0 = __builtin_amdgcn_wmma_f32_16x16x32_bf16(
        false, a[cur].v, false, b0[cur].v, (short)0, acc0, false, false);
    acc1 = __builtin_amdgcn_wmma_f32_16x16x32_bf16(
        false, a[cur].v, false, b1[cur].v, (short)0, acc1, false, false);
    acc2 = __builtin_amdgcn_wmma_f32_16x16x32_bf16(
        false, a[cur].v, false, b2[cur].v, (short)0, acc2, false, false);
    acc3 = __builtin_amdgcn_wmma_f32_16x16x32_bf16(
        false, a[cur].v, false, b3[cur].v, (short)0, acc3, false, false);
  }
  // D layout: vgpr r -> M = r + 8*(lane>>4), N = lane&15
  int n = lane & 15, hi = lane >> 4;
  float bv0 = bias[ngrp * 64 + n];
  float bv1 = bias[ngrp * 64 + 16 + n];
  float bv2 = bias[ngrp * 64 + 32 + n];
  float bv3 = bias[ngrp * 64 + 48 + n];
  bf16* orow = out + ngrp * 64 + n;
#pragma unroll
  for (int r = 0; r < 8; ++r) {
    size_t off = (size_t)(mtile * 16 + r + hi * 8) * 256;
    orow[off]      = (bf16)((acc0[r] + bv0) * scale);
    orow[off + 16] = (bf16)((acc1[r] + bv1) * scale);
    orow[off + 32] = (bf16)((acc2[r] + bv2) * scale);
    orow[off + 48] = (bf16)((acc3[r] + bv3) * scale);
  }
}

// ---------------------------------------------------------------------------
// Scores + softmax + (head,L) accumulation. Block = (batch b, head h, L-tile).
// 4 waves split the 25 S-tiles; head dim 32 == WMMA K, so one WMMA per tile.
// ---------------------------------------------------------------------------
__global__ void __launch_bounds__(128) scores_kernel(
    const bf16* __restrict__ Qp, const bf16* __restrict__ Kp,
    float* __restrict__ aw) {
  __shared__ float sc[16][404];       // padded: conflict-free row stride
  __shared__ float pmax[16][8];
  __shared__ float psum[16][8];
  __shared__ float rowmax_s[16];
  __shared__ float rowinv_s[16];

  int ltile = blockIdx.x;             // 0..24
  int head  = blockIdx.y;             // 0..7
  int b     = blockIdx.z;             // 0..63
  int lane = threadIdx.x & 31, wave = threadIdx.x >> 5;
  int half = lane >> 4, mr = lane & 15;

  // A fragment (Q rows), reused by every S-tile of this wave
  Frag a;
  const bf16* arow = Qp + (size_t)(b * 400 + ltile * 16 + mr) * 256 + head * 32;
  const uint4* ap = (const uint4*)(arow + half * 16);
  a.u[0] = ap[0];
  a.u[1] = ap[1];

  for (int st = wave; st < 25; st += 4) {
    Frag bm;
    const bf16* brow =
        Kp + (size_t)(b * 400 + st * 16 + mr) * 256 + head * 32 + half * 16;
    const uint4* bp = (const uint4*)brow;
    bm.u[0] = bp[0];
    bm.u[1] = bp[1];
    v8f acc = {};
    acc = __builtin_amdgcn_wmma_f32_16x16x32_bf16(
        false, a.v, false, bm.v, (short)0, acc, false, false);
#pragma unroll
    for (int r = 0; r < 8; ++r)
      sc[r + half * 8][st * 16 + mr] = acc[r];
  }
  __syncthreads();

  // Softmax over 400 cols per row; 8 threads x 50 cols per row.
  int row = threadIdx.x & 15;
  int chunk = threadIdx.x >> 4;       // 0..7
  int c0 = chunk * 50;
  float mx = -1e30f;
  for (int c = 0; c < 50; ++c) mx = fmaxf(mx, sc[row][c0 + c]);
  pmax[row][chunk] = mx;
  __syncthreads();
  if (threadIdx.x < 16) {
    float m = pmax[threadIdx.x][0];
#pragma unroll
    for (int k = 1; k < 8; ++k) m = fmaxf(m, pmax[threadIdx.x][k]);
    rowmax_s[threadIdx.x] = m;
  }
  __syncthreads();
  float rm = rowmax_s[row];
  float s = 0.0f;
  for (int c = 0; c < 50; ++c) {
    float e = __expf(sc[row][c0 + c] - rm);
    sc[row][c0 + c] = e;
    s += e;
  }
  psum[row][chunk] = s;
  __syncthreads();
  if (threadIdx.x < 16) {
    float t = 0.0f;
#pragma unroll
    for (int k = 0; k < 8; ++k) t += psum[threadIdx.x][k];
    rowinv_s[threadIdx.x] = 1.0f / t;
  }
  __syncthreads();
  // Column sums of probabilities over the 16 L-rows of this tile.
  for (int col = threadIdx.x; col < 400; col += 128) {
    float accv = 0.0f;
#pragma unroll
    for (int m = 0; m < 16; ++m) accv += sc[m][col] * rowinv_s[m];
    atomicAdd(&aw[b * 400 + col], accv);
  }
}

// ---------------------------------------------------------------------------
// Final gating + re-tiling + merge with 0.5*upsampled global.
// ---------------------------------------------------------------------------
__global__ void final_kernel(const float* __restrict__ x, const float* __restrict__ upg,
                             const float* __restrict__ aw,
                             const float* __restrict__ salw,
                             const float* __restrict__ salb,
                             float* __restrict__ out) {
  int idx = blockIdx.x * 256 + threadIdx.x;           // 256*160*160
  int c = idx / 25600, rem = idx % 25600;
  int Y = rem / 160, X = rem % 160;
  int bi = Y / 80, bj = X / 80, yy = Y % 80, xx = X % 80;
  int b = bi * 2 + bj;
  int i = yy / 20, py = yy % 20, j = xx / 20, px = xx % 20;
  int patch = b * 16 + i * 4 + j;
  float am = aw[patch * 400 + py * 20 + px] * (1.0f / 3200.0f);  // /8 heads /400 L
  float z = salw[0] * am + salb[0];
  float imp = 0.6f / (1.0f + __expf(-z));
  float locv = x[((size_t)b * 256 + c) * 6400 + yy * 80 + xx];
  out[idx] = locv * imp + 0.5f * upg[idx];
}

// ---------------------------------------------------------------------------
extern "C" void kernel_launch(void* const* d_in, const int* in_sizes, int n_in,
                              void* d_out, int out_size, void* d_ws, size_t ws_size,
                              hipStream_t stream) {
  (void)in_sizes; (void)n_in; (void)out_size; (void)ws_size;
  const float* x    = (const float*)d_in[0];   // [5,256,80,80]
  const float* ipw  = (const float*)d_in[1];   // [768,256]
  const float* ipb  = (const float*)d_in[2];   // [768]
  const float* salw = (const float*)d_in[3];   // [1]
  const float* salb = (const float*)d_in[4];   // [1]
  float* out = (float*)d_out;                  // [1,256,160,160]

  char* ws = (char*)d_ws;
  float* upg = (float*)(ws);                   // 26,214,400 B  f32 [256,160,160]
  float* pos = (float*)(ws + 26214400);        //    409,600 B  f32 [400,256]
  float* aw  = (float*)(ws + 26624000);        //    102,400 B  f32 [64,400]
  bf16*  wqk = (bf16*)(ws + 26726400);         //    262,144 B  bf16 [512,256]
  bf16*  qin = (bf16*)(ws + 26988544);         // 13,107,200 B  bf16 [25600,256]
  bf16*  kin = (bf16*)(ws + 40095744);         // 13,107,200 B
  bf16*  Qp  = (bf16*)(ws + 53202944);         // 13,107,200 B
  bf16*  Kp  = (bf16*)(ws + 66310144);         // 13,107,200 B

  hipMemsetAsync(aw, 0, 64 * 400 * sizeof(float), stream);
  pos_kernel  <<<400,   256, 0, stream>>>(pos);
  upglb_kernel<<<25600, 256, 0, stream>>>(x, upg);
  wcvt_kernel <<<512,   256, 0, stream>>>(ipw, wqk);
  qkin_kernel <<<25600, 256, 0, stream>>>(x, upg, pos, qin, kin);
  proj_kernel <<<dim3(4, 400, 2), 128, 0, stream>>>(qin, kin, wqk, ipb, Qp, Kp);
  scores_kernel<<<dim3(25, 8, 64), 128, 0, stream>>>(Qp, Kp, aw);
  final_kernel<<<25600, 256, 0, stream>>>(x, upg, aw, salw, salb, out);
}
